// RPPO_E2E_VAT_5617817223177
// MI455X (gfx1250) — matrix operand order repaired and checked by reading, not compiled, not run on hardware
//
#include <hip/hip_runtime.h>
#include <hip/hip_bf16.h>

// RPPO actor-critic forward for MI455X (gfx1250, wave32, WMMA).
// Pipeline: fused conv1(a+c, x read once) -> conv2 -> FC -> gi GEMM ->
// persistent GRU scan (WMMA per step) -> MLP -> softmax/entropy/value head.
// All dense math: v_wmma_f32_16x16x32_f16 (f16 in, f32 accum).
// GEMMs double-buffer LDS so global-load latency hides behind WMMA.

typedef __attribute__((ext_vector_type(16))) _Float16 v16h;
typedef __attribute__((ext_vector_type(8)))  _Float16 v8h;
typedef __attribute__((ext_vector_type(4)))  _Float16 v4h;
typedef __attribute__((ext_vector_type(2)))  _Float16 v2h;
typedef __attribute__((ext_vector_type(8)))  float    v8f;

#define DEV __device__ __forceinline__

DEV v16h combine8(v8h lo, v8h hi) {
  v16h r;
#pragma unroll
  for (int i = 0; i < 8; ++i) { r[i] = lo[i]; r[8 + i] = hi[i]; }
  return r;
}

// A fragment (16xK=32, f16) from row-major [16][ld] tile (CDNA5 ISA 7.12.2).
DEV v16h afrag(const _Float16* base, int ld, int k0) {
  const int lane = threadIdx.x & 31;
  const _Float16* p = base + (lane & 15) * ld + k0 + ((lane >> 4) << 4);
  return combine8(*(const v8h*)p, *(const v8h*)(p + 8));
}

// B fragment (K=32 x 16, f16); weights stored [N=16 rows][ld], B(k,n)=Wrow[n][k].
DEV v16h bfrag(const _Float16* base, int ld, int k0) {
  const int lane = threadIdx.x & 31;
  const _Float16* p = base + (lane & 15) * ld + k0 + ((lane >> 4) << 4);
  return combine8(*(const v8h*)p, *(const v8h*)(p + 8));
}

// Same as bfrag but straight from global f16 weights [N][ld].
DEV v16h bfrag_g(const _Float16* W, int ld, int n0, int k0) {
  const int lane = threadIdx.x & 31;
  const _Float16* p = W + (size_t)(n0 + (lane & 15)) * ld + k0 + ((lane >> 4) << 4);
  return combine8(*(const v8h*)p, *(const v8h*)(p + 8));
}

DEV v8f wmma16(v16h a, v16h b, v8f c) {
  return __builtin_amdgcn_wmma_f32_16x16x32_f16(false, a, false, b, (short)0, c,
                                                false, false);
}

// ---------------------------------------------------------------------------
// conv1 for BOTH branches fused: x is 462MB (> L2), read it exactly once.
// Implicit GEMM: M = 16 output pixels/tile, N = 16 out channels, K = 4*8*8=256.
// im2col rows are kw-contiguous in x -> float4 loads (84 % 4 == 0 keeps 16B align).
__global__ __launch_bounds__(128)
void conv1_fused(const float* __restrict__ x,
                 const float* __restrict__ wA, const float* __restrict__ bA,
                 const float* __restrict__ wC, const float* __restrict__ bC,
                 _Float16* __restrict__ outA, _Float16* __restrict__ outC) {
  __shared__ _Float16 As[4][16][264];
  __shared__ _Float16 Ws[2][16][264];
  const int tid = threadIdx.x, wave = tid >> 5, lane = tid & 31;

  for (int e = tid; e < 1024; e += 128) {           // weights, float4 wide
    int r = e >> 6, k = (e & 63) * 4;
    const float4 fa = *(const float4*)&wA[e * 4];
    const float4 fc = *(const float4*)&wC[e * 4];
    v4h ha, hc;
    ha[0] = (_Float16)fa.x; ha[1] = (_Float16)fa.y;
    ha[2] = (_Float16)fa.z; ha[3] = (_Float16)fa.w;
    hc[0] = (_Float16)fc.x; hc[1] = (_Float16)fc.y;
    hc[2] = (_Float16)fc.z; hc[3] = (_Float16)fc.w;
    *(v4h*)&Ws[0][r][k] = ha;
    *(v4h*)&Ws[1][r][k] = hc;
  }
  const int gtile = blockIdx.x * 4 + wave;          // 0..102399
  const int n = gtile / 25, tstart = (gtile % 25) * 16;
  for (int e = lane; e < 1024; e += 32) {           // im2col, float4 wide
    int r = e >> 6, q = e & 63;                     // k = q*4
    int p = tstart + r, oh = p / 20, ow = p % 20;
    int c = q >> 4, kh = (q >> 1) & 7, kw0 = (q & 1) * 4;
    const float4 f =
        *(const float4*)&x[(((size_t)n * 4 + c) * 84 + oh * 4 + kh) * 84 + ow * 4 + kw0];
    v4h h;
    h[0] = (_Float16)f.x; h[1] = (_Float16)f.y;
    h[2] = (_Float16)f.z; h[3] = (_Float16)f.w;
    *(v4h*)&As[wave][r][q * 4] = h;
  }
  __syncthreads();

  v8f accA = {}, accC = {};
#pragma unroll
  for (int k0 = 0; k0 < 256; k0 += 32) {
    v16h a = afrag(&As[wave][0][0], 264, k0);
    accA = wmma16(a, bfrag(&Ws[0][0][0], 264, k0), accA);
    accC = wmma16(a, bfrag(&Ws[1][0][0], 264, k0), accC);
  }
  const int ch = lane & 15;
  const int rb = tstart + ((lane >> 4) << 3);
  const float biasA = bA[ch], biasC = bC[ch];
#pragma unroll
  for (int r = 0; r < 8; ++r) {
    size_t oi = ((size_t)n * 16 + ch) * 400 + (rb + r);
    outA[oi] = (_Float16)fmaxf(accA[r] + biasA, 0.f);
    outC[oi] = (_Float16)fmaxf(accC[r] + biasC, 0.f);
  }
}

// ---------------------------------------------------------------------------
// conv2 implicit GEMM: M = 16 of 81 pixels (6 tiles, padded), N = 32, K = 256.
__global__ __launch_bounds__(128)
void conv2_wmma(const _Float16* __restrict__ in,   // [4096][16][400]
                const float* __restrict__ w,       // [32][256]
                const float* __restrict__ bias,    // [32]
                _Float16* __restrict__ out) {      // [4096][2592], ch*81+p
  __shared__ _Float16 As[4][16][264];
  __shared__ _Float16 Ws[32][264];
  const int tid = threadIdx.x, wave = tid >> 5, lane = tid & 31;

  for (int e = tid; e < 2048; e += 128) {           // weights, float4 wide
    int r = e >> 6, k = (e & 63) * 4;
    const float4 f = *(const float4*)&w[e * 4];
    v4h h;
    h[0] = (_Float16)f.x; h[1] = (_Float16)f.y;
    h[2] = (_Float16)f.z; h[3] = (_Float16)f.w;
    *(v4h*)&Ws[r][k] = h;
  }
  const int gtile = blockIdx.x * 4 + wave;          // 0..24575
  const int n = gtile / 6, tstart = (gtile % 6) * 16;
  for (int e = lane; e < 2048; e += 32) {           // im2col, v2h (4B) wide
    int r = e >> 7, q = e & 127;                    // k = q*2
    int p = tstart + r;
    v2h v; v[0] = (_Float16)0.f; v[1] = (_Float16)0.f;
    if (p < 81) {
      int oh = p / 9, ow = p % 9;
      int c = q >> 3, kh = (q >> 1) & 3, kw0 = (q & 1) * 2;
      v = *(const v2h*)&in[((size_t)n * 16 + c) * 400 + (oh * 2 + kh) * 20 + ow * 2 + kw0];
    }
    *(v2h*)&As[wave][r][q * 2] = v;
  }
  __syncthreads();

  v8f acc0 = {}, acc1 = {};
#pragma unroll
  for (int k0 = 0; k0 < 256; k0 += 32) {
    v16h a = afrag(&As[wave][0][0], 264, k0);
    acc0 = wmma16(a, bfrag(&Ws[0][0], 264, k0), acc0);
    acc1 = wmma16(a, bfrag(&Ws[16][0], 264, k0), acc1);
  }
  const int rb = tstart + ((lane >> 4) << 3);
#pragma unroll
  for (int j = 0; j < 2; ++j) {
    int ch = j * 16 + (lane & 15);
    float bv = bias[ch];
    v8f acc = j ? acc1 : acc0;
#pragma unroll
    for (int r = 0; r < 8; ++r) {
      int p = rb + r;
      if (p < 81)
        out[(size_t)n * 2592 + ch * 81 + p] = (_Float16)fmaxf(acc[r] + bv, 0.f);
    }
  }
}

// ---------------------------------------------------------------------------
// Generic WMMA GEMM: C[M,N] = act(A[M,K](f16) @ W[N,K](f32)^T + bias).
// Block = 128 thr (4 waves), tile = 64x64, K-step 32, double-buffered LDS:
// global loads for tile k+1 are issued before the 16 WMMAs of tile k and only
// waited on at the LDS commit afterwards, hiding global latency behind math.
DEV void gemm_issue(const _Float16* __restrict__ A, int lda,
                    const float* __restrict__ W, int Ks,
                    int bm, int bn, int M, int N, int K, int k0, int tid,
                    v8h ar[2], float4 wr[4]) {
#pragma unroll
  for (int i = 0; i < 2; ++i) {                    // A: 64x32 halves
    int id = tid + i * 128, r = id >> 2, c = (id & 3) * 8;
    int gr = bm + r, gc = k0 + c;
    v8h v;
#pragma unroll
    for (int t = 0; t < 8; ++t) v[t] = (_Float16)0.f;
    if (gr < M) {
      if (gc + 8 <= K) v = *(const v8h*)&A[(size_t)gr * lda + gc];
      else
        for (int t = 0; t < 8; ++t)
          if (gc + t < K) v[t] = A[(size_t)gr * lda + gc + t];
    }
    ar[i] = v;
  }
#pragma unroll
  for (int i = 0; i < 4; ++i) {                    // W: 64x32 floats
    int id = tid + i * 128, r = id >> 3, c = (id & 7) * 4;
    int gr = bn + r, gc = k0 + c;
    float4 v = make_float4(0.f, 0.f, 0.f, 0.f);
    if (gr < N) {
      if (gc + 4 <= K) v = *(const float4*)&W[(size_t)gr * Ks + gc];
      else {
        const float* p = &W[(size_t)gr * Ks];
        if (gc + 0 < K) v.x = p[gc + 0];
        if (gc + 1 < K) v.y = p[gc + 1];
        if (gc + 2 < K) v.z = p[gc + 2];
        if (gc + 3 < K) v.w = p[gc + 3];
      }
    }
    wr[i] = v;
  }
}

DEV void gemm_commit(_Float16* __restrict__ As, _Float16* __restrict__ Ws,
                     int tid, const v8h ar[2], const float4 wr[4]) {
#pragma unroll
  for (int i = 0; i < 2; ++i) {
    int id = tid + i * 128, r = id >> 2, c = (id & 3) * 8;
    *(v8h*)&As[r * 40 + c] = ar[i];
  }
#pragma unroll
  for (int i = 0; i < 4; ++i) {
    int id = tid + i * 128, r = id >> 3, c = (id & 7) * 4;
    v4h h;
    h[0] = (_Float16)wr[i].x; h[1] = (_Float16)wr[i].y;
    h[2] = (_Float16)wr[i].z; h[3] = (_Float16)wr[i].w;
    *(v4h*)&Ws[r * 40 + c] = h;
  }
}

__global__ __launch_bounds__(128)
void gemm_wmma_k(const _Float16* __restrict__ A, int lda,
                 const float* __restrict__ W, const float* __restrict__ bias,
                 _Float16* __restrict__ Ch, float* __restrict__ Cf, int ldc,
                 int M, int N, int K, int relu) {
  __shared__ _Float16 As[2][64][40];
  __shared__ _Float16 Ws[2][64][40];
  const int tid = threadIdx.x, wave = tid >> 5, lane = tid & 31;
  const int bm = blockIdx.y * 64, bn = blockIdx.x * 64;
  const int nk = (K + 31) >> 5;
  v8f acc[4] = {};
  v8h ar[2]; float4 wr[4];

  gemm_issue(A, lda, W, K, bm, bn, M, N, K, 0, tid, ar, wr);
  gemm_commit(&As[0][0][0], &Ws[0][0][0], tid, ar, wr);
  __syncthreads();

  for (int kt = 0; kt < nk; ++kt) {
    const int cur = kt & 1;
    const bool more = (kt + 1) < nk;
    if (more)
      gemm_issue(A, lda, W, K, bm, bn, M, N, K, (kt + 1) * 32, tid, ar, wr);
    if ((kt + 2) * 32 < K) {   // CDNA5 global_prefetch_b8 of the tile after next
      __builtin_prefetch(&A[(size_t)(bm + (tid >> 1)) * lda + (kt + 2) * 32], 0, 1);
      __builtin_prefetch(&W[(size_t)(bn + (tid >> 1)) * K + (kt + 2) * 32], 0, 1);
    }
    v16h a = afrag(&As[cur][wave * 16][0], 40, 0);
#pragma unroll
    for (int j = 0; j < 4; ++j)
      acc[j] = wmma16(a, bfrag(&Ws[cur][j * 16][0], 40, 0), acc[j]);
    if (more) gemm_commit(&As[cur ^ 1][0][0], &Ws[cur ^ 1][0][0], tid, ar, wr);
    __syncthreads();
  }

#pragma unroll
  for (int j = 0; j < 4; ++j) {
    int col = bn + j * 16 + (lane & 15);
    if (col >= N) continue;
    float bv = bias ? bias[col] : 0.f;
    int rb = bm + wave * 16 + ((lane >> 4) << 3);
#pragma unroll
    for (int r = 0; r < 8; ++r) {
      int row = rb + r;
      if (row < M) {
        float v = acc[j][r] + bv;
        if (relu) v = fmaxf(v, 0.f);
        if (Ch) Ch[(size_t)row * ldc + col] = (_Float16)v;
        if (Cf) Cf[(size_t)row * ldc + col] = v;
      }
    }
  }
}

// ---------------------------------------------------------------------------
__global__ void f32_to_f16(const float* __restrict__ in,
                           _Float16* __restrict__ out, int n) {
  int i = blockIdx.x * blockDim.x + threadIdx.x;
  if (i < n) out[i] = (_Float16)in[i];
}

// ---------------------------------------------------------------------------
// Persistent GRU scan: grid=2 (branch), block=512 (16 waves). gi precomputed.
// Per step: gh = h @ Whh^T via 96 WMMA tiles (2 Mtiles x 48 Ntiles), gates in
// VALU, h kept resident in LDS. Whh streamed f16 from L2 (393KB resident).
__global__ __launch_bounds__(512)
void gru_scan(const _Float16* __restrict__ whhA, const _Float16* __restrict__ whhC,
              const float* __restrict__ giA, const float* __restrict__ giC,
              const float* __restrict__ bhhA, const float* __restrict__ bhhC,
              const float* __restrict__ h0,   // [2][32][256]
              const float* __restrict__ done, // [4096]
              _Float16* __restrict__ hsA, _Float16* __restrict__ hsC,
              float* __restrict__ hT) {       // [2][32][256]
  const int br = blockIdx.x;
  const _Float16* whh = br ? whhC : whhA;
  const float* gi = br ? giC : giA;
  const float* bhh = br ? bhhC : bhhA;
  _Float16* hs = br ? hsC : hsA;
  __shared__ _Float16 sh[32 * 256];   // hidden state, f16
  __shared__ _Float16 sgh[32 * 768];  // h @ Whh^T gates
  const int tid = threadIdx.x, wave = tid >> 5, lane = tid & 31;

  for (int e = tid; e < 8192; e += 512) sh[e] = (_Float16)h0[br * 8192 + e];
  __syncthreads();

  for (int t = 0; t < 128; ++t) {
    for (int e = tid; e < 8192; e += 512) {            // done-mask reset
      float d = done[t * 32 + (e >> 8)];
      sh[e] = (_Float16)((1.f - d) * (float)sh[e]);
    }
    __syncthreads();

#pragma unroll
    for (int u = 0; u < 6; ++u) {                      // 96 tiles / 16 waves
      int ti = wave * 6 + u, mt = ti / 48, ct = ti % 48;
      v8f acc = {};
#pragma unroll
      for (int k0 = 0; k0 < 256; k0 += 32)
        acc = wmma16(afrag(sh + mt * 16 * 256, 256, k0),
                     bfrag_g(whh, 256, ct * 16, k0), acc);
      int col = ct * 16 + (lane & 15);
      int rb = mt * 16 + ((lane >> 4) << 3);
#pragma unroll
      for (int r = 0; r < 8; ++r) sgh[(rb + r) * 768 + col] = (_Float16)acc[r];
    }
    __syncthreads();

    for (int e = tid; e < 8192; e += 512) {            // gate math
      int b = e >> 8, j = e & 255;
      const float* g = gi + ((size_t)t * 32 + b) * 768;
      float hr = (float)sgh[b * 768 + j] + bhh[j];
      float hz = (float)sgh[b * 768 + 256 + j] + bhh[256 + j];
      float hn = (float)sgh[b * 768 + 512 + j] + bhh[512 + j];
      float rg = 1.f / (1.f + __expf(-(g[j] + hr)));
      float zg = 1.f / (1.f + __expf(-(g[256 + j] + hz)));
      float ng = tanhf(g[512 + j] + rg * hn);
      float hnew = (1.f - zg) * ng + zg * (float)sh[e];
      sh[e] = (_Float16)hnew;
      hs[((size_t)t * 32 + b) * 256 + j] = (_Float16)fmaxf(hnew, 0.f);
    }
    __syncthreads();
  }
  for (int e = tid; e < 8192; e += 512) hT[br * 8192 + e] = (float)sh[e];
}

// ---------------------------------------------------------------------------
// Tiny heads: logits (7), log-softmax, logp gather, entropy, value.
__global__ void head_kernel(const _Float16* __restrict__ h2a,
                            const _Float16* __restrict__ h2c,
                            const float* __restrict__ aw3, const float* __restrict__ ab3,
                            const float* __restrict__ cw3, const float* __restrict__ cb3,
                            const int* __restrict__ action,
                            float* __restrict__ out) {
  int row = blockIdx.x * blockDim.x + threadIdx.x;
  if (row >= 4096) return;
  const _Float16* h = h2a + (size_t)row * 100;
  float logits[7];
  float mx = -3.0e38f;
#pragma unroll
  for (int o = 0; o < 7; ++o) {
    float s = ab3[o];
    for (int k = 0; k < 100; ++k) s += (float)h[k] * aw3[o * 100 + k];
    logits[o] = s;
    mx = fmaxf(mx, s);
  }
  float se = 0.f;
#pragma unroll
  for (int o = 0; o < 7; ++o) se += __expf(logits[o] - mx);
  float lse = __logf(se) + mx;
  float ent = 0.f;
#pragma unroll
  for (int o = 0; o < 7; ++o) {
    float lp = logits[o] - lse;
    ent -= __expf(lp) * lp;
  }
  int act = action[row];
  const _Float16* hc = h2c + (size_t)row * 100;
  float v = cb3[0];
  for (int k = 0; k < 100; ++k) v += (float)hc[k] * cw3[k];
  ((int*)out)[row] = act;         // action passthrough (int bits)
  out[4096 + row] = logits[act] - lse;
  out[8192 + row] = ent;
  out[12288 + row] = v;
}

// ---------------------------------------------------------------------------
extern "C" void kernel_launch(void* const* d_in, const int* in_sizes, int n_in,
                              void* d_out, int out_size, void* d_ws, size_t ws_size,
                              hipStream_t stream) {
  (void)in_sizes; (void)n_in; (void)out_size; (void)ws_size;
  const float* x    = (const float*)d_in[0];
  const float* gru0 = (const float*)d_in[1];
  const float* done = (const float*)d_in[2];
  const int*   act  = (const int*)d_in[3];
  // per-branch param order: 0 c1w 1 c1b 2 c2w 3 c2b 4 fw 5 fb 6 wih 7 whh
  //                         8 bih 9 bhh 10 h1w 11 h1b 12 h2w 13 h2b 14 h3w 15 h3b
  const float* A[16]; const float* C[16];
  for (int i = 0; i < 16; ++i) {
    A[i] = (const float*)d_in[4 + i];
    C[i] = (const float*)d_in[20 + i];
  }

  char* wsp = (char*)d_ws; size_t off = 0;
  auto take = [&](size_t bytes) {
    void* p = wsp + off;
    off = (off + bytes + 255) & ~(size_t)255;
    return p;
  };
  _Float16* c1a  = (_Float16*)take(4096ull * 16 * 400 * 2);
  _Float16* c1c  = (_Float16*)take(4096ull * 16 * 400 * 2);
  _Float16* y2a  = (_Float16*)take(4096ull * 2592 * 2);
  _Float16* y2c  = (_Float16*)take(4096ull * 2592 * 2);
  _Float16* fta  = (_Float16*)take(4096ull * 256 * 2);
  _Float16* ftc  = (_Float16*)take(4096ull * 256 * 2);
  float*    gia  = (float*)take(4096ull * 768 * 4);
  float*    gic  = (float*)take(4096ull * 768 * 4);
  _Float16* whha = (_Float16*)take(768ull * 256 * 2);
  _Float16* whhc = (_Float16*)take(768ull * 256 * 2);
  _Float16* hsa  = (_Float16*)take(4096ull * 256 * 2);
  _Float16* hsc  = (_Float16*)take(4096ull * 256 * 2);
  _Float16* h1a  = (_Float16*)take(4096ull * 200 * 2);
  _Float16* h1c  = (_Float16*)take(4096ull * 200 * 2);
  _Float16* h2a  = (_Float16*)take(4096ull * 100 * 2);
  _Float16* h2c  = (_Float16*)take(4096ull * 100 * 2);
  float* out = (float*)d_out;

  // conv1 (both branches, x read once)
  conv1_fused<<<dim3(25600), dim3(128), 0, stream>>>(x, A[0], A[1], C[0], C[1],
                                                     c1a, c1c);
  // conv2
  conv2_wmma<<<dim3(6144), dim3(128), 0, stream>>>(c1a, A[2], A[3], y2a);
  conv2_wmma<<<dim3(6144), dim3(128), 0, stream>>>(c1c, C[2], C[3], y2c);
  // FC: feats = relu(y2 @ fw^T + fb), M=4096 N=256 K=2592
  gemm_wmma_k<<<dim3(4, 64), dim3(128), 0, stream>>>(y2a, 2592, A[4], A[5], fta,
                                                     nullptr, 256, 4096, 256, 2592, 1);
  gemm_wmma_k<<<dim3(4, 64), dim3(128), 0, stream>>>(y2c, 2592, C[4], C[5], ftc,
                                                     nullptr, 256, 4096, 256, 2592, 1);
  // gi = feats @ wih^T + bih (hoisted out of the scan), M=4096 N=768 K=256
  gemm_wmma_k<<<dim3(12, 64), dim3(128), 0, stream>>>(fta, 256, A[6], A[8], nullptr,
                                                      gia, 768, 4096, 768, 256, 0);
  gemm_wmma_k<<<dim3(12, 64), dim3(128), 0, stream>>>(ftc, 256, C[6], C[8], nullptr,
                                                      gic, 768, 4096, 768, 256, 0);
  // whh -> f16 once, streamed from L2 inside the scan
  f32_to_f16<<<dim3((196608 + 255) / 256), dim3(256), 0, stream>>>(A[7], whha, 196608);
  f32_to_f16<<<dim3((196608 + 255) / 256), dim3(256), 0, stream>>>(C[7], whhc, 196608);
  // sequential GRU scan, persistent block per branch; hT -> tail of d_out
  gru_scan<<<dim3(2), dim3(512), 0, stream>>>(whha, whhc, gia, gic, A[9], C[9],
                                              gru0, done, hsa, hsc, out + 16384);
  // MLP: 256 -> 200 -> 100
  gemm_wmma_k<<<dim3(4, 64), dim3(128), 0, stream>>>(hsa, 256, A[10], A[11], h1a,
                                                     nullptr, 200, 4096, 200, 256, 1);
  gemm_wmma_k<<<dim3(4, 64), dim3(128), 0, stream>>>(hsc, 256, C[10], C[11], h1c,
                                                     nullptr, 200, 4096, 200, 256, 1);
  gemm_wmma_k<<<dim3(2, 64), dim3(128), 0, stream>>>(h1a, 200, A[12], A[13], h2a,
                                                     nullptr, 100, 4096, 100, 200, 1);
  gemm_wmma_k<<<dim3(2, 64), dim3(128), 0, stream>>>(h1c, 200, C[12], C[13], h2c,
                                                     nullptr, 100, 4096, 100, 200, 1);
  // heads: logits/log-softmax/entropy + value
  head_kernel<<<dim3(16), dim3(256), 0, stream>>>(h2a, h2c, A[14], A[15], C[14],
                                                  C[15], act, out);
}
